// DATTD_31044023615504
// MI455X (gfx1250) — matrix operand order repaired and checked by reading, not compiled, not run on hardware
//
#include <hip/hip_runtime.h>
#include <hip/hip_bf16.h>

typedef _Float16 half_t;
typedef __attribute__((ext_vector_type(16))) _Float16 v16h;
typedef __attribute__((ext_vector_type(8)))  _Float16 h8;
typedef __attribute__((ext_vector_type(8)))  float    v8f;
typedef unsigned int v4u __attribute__((ext_vector_type(4)));
typedef int          v8i __attribute__((ext_vector_type(8)));
typedef int          v4i __attribute__((ext_vector_type(4)));

#define CDIM   384
#define NHEADS 8
#define HCH    48
#define BATCH  2
#define HDIM   192
#define WDIM   192
#define HWSZ   (HDIM*WDIM)      /* 36864 */
#define PCOLS  (BATCH*HWSZ)     /* 73728 */

// ---------------------------------------------------------------------------
// WMMA fragment helpers (wave32, 16-bit A 16x32 / B 32x16 / C 16x16 f32)
// ---------------------------------------------------------------------------
__device__ __forceinline__ v16h load_frag_a(const half_t* __restrict__ src, int ld, int lane) {
    int m = lane & 15;
    int s = (lane >> 4) * 8;
    const half_t* r = src + (size_t)m * ld;
    h8 lo = *(const h8*)(r + s);
    h8 hi = *(const h8*)(r + 16 + s);
    v16h f;
#pragma unroll
    for (int i = 0; i < 8; ++i) { f[i] = lo[i]; f[i + 8] = hi[i]; }
    return f;
}

__device__ __forceinline__ v16h load_frag_bt(const half_t* __restrict__ srcT, int ldk, int lane) {
    int n  = lane & 15;
    int ko = (lane >> 4) * 8;
    const half_t* r = srcT + (size_t)n * ldk + ko;
    h8 lo = *(const h8*)(r);
    h8 hi = *(const h8*)(r + 16);
    v16h f;
#pragma unroll
    for (int i = 0; i < 8; ++i) { f[i] = lo[i]; f[i + 8] = hi[i]; }
    return f;
}

__device__ __forceinline__ v8f wmma_f16(v16h a, v16h b, v8f c) {
    return __builtin_amdgcn_wmma_f32_16x16x32_f16(false, a, false, b, (short)0, c, false, false);
}

// ---------------------------------------------------------------------------
// TDM: 2D tile (tile_d0 contiguous elems x tile_d1 rows, 2-byte data) -> LDS
// D# packing per CDNA5 ISA
// ---------------------------------------------------------------------------
__device__ __forceinline__ void tdm_load_2d(unsigned int lds_off, const void* gaddr,
                                            unsigned int tile_d0, unsigned int tile_d1,
                                            unsigned int tensor_d0, unsigned int tensor_d1,
                                            unsigned int stride_e) {
    unsigned long long ga = (unsigned long long)(uintptr_t)gaddr;
    v4u g0;
    g0[0] = 1u;                                   // count=1, user descriptor
    g0[1] = lds_off;                              // lds_addr (bytes)
    g0[2] = (unsigned int)ga;                     // global_addr[31:0]
    g0[3] = (unsigned int)((ga >> 32) & 0x1FFFFFFu) | (2u << 30);   // addr[56:32] | type=2
    v8i g1;
    g1[0] = (int)(1u << 16);                      // data_size=1 -> 2 bytes
    g1[1] = (int)((tensor_d0 & 0xFFFFu) << 16);                         // tensor_dim0[15:0]
    g1[2] = (int)(((tensor_d0 >> 16) & 0xFFFFu) | ((tensor_d1 & 0xFFFFu) << 16));
    g1[3] = (int)(((tensor_d1 >> 16) & 0xFFFFu) | ((tile_d0 & 0xFFFFu) << 16));
    g1[4] = (int)(tile_d1 & 0xFFFFu);             // tile_dim1, tile_dim2=0
    g1[5] = (int)stride_e;                        // tensor_dim0_stride[31:0]
    g1[6] = 0;                                    // stride hi, dim1_stride lo
    g1[7] = 0;
    v4i z4 = {0, 0, 0, 0};
#if defined(__clang_major__) && __clang_major__ >= 23
    v8i z8 = {0, 0, 0, 0, 0, 0, 0, 0};
    __builtin_amdgcn_tensor_load_to_lds(g0, g1, z4, z4, z8, 0);
#else
    __builtin_amdgcn_tensor_load_to_lds(g0, g1, z4, z4, 0);
#endif
}

// ---------------------------------------------------------------------------
// 0) convert the four 384x384 fp32 weight matrices to f16 (L2-resident)
// ---------------------------------------------------------------------------
__global__ __launch_bounds__(256) void conv_w16(
    const float* __restrict__ wq, const float* __restrict__ wk,
    const float* __restrict__ wv, const float* __restrict__ wf,
    half_t* __restrict__ out)
{
    long gid = (long)blockIdx.x * 256 + threadIdx.x;
    int z = (int)(gid / (CDIM * CDIM));
    int i = (int)(gid % (CDIM * CDIM));
    const float* src = (z == 0) ? wq : (z == 1) ? wk : (z == 2) ? wv : wf;
    out[gid] = (half_t)src[i];
}

// ---------------------------------------------------------------------------
// 1) LayerNorm over channels (per pixel), fp32 BCHW -> f16 K-major [P][C]
// ---------------------------------------------------------------------------
__global__ __launch_bounds__(256) void ln_kernel(
    const float* __restrict__ q, const float* __restrict__ k, const float* __restrict__ v,
    const float* __restrict__ wq, const float* __restrict__ bq,
    const float* __restrict__ wk, const float* __restrict__ bk,
    const float* __restrict__ wv, const float* __restrict__ bv,
    half_t* __restrict__ out)
{
    long gid = (long)blockIdx.x * 256 + threadIdx.x;
    int z = (int)(gid / PCOLS);
    int p = (int)(gid % PCOLS);
    int b = p / HWSZ, hw = p % HWSZ;
    const float* x = (z == 0) ? q : (z == 1) ? k : v;
    const float* w = (z == 0) ? wq : (z == 1) ? wk : wv;
    const float* bb = (z == 0) ? bq : (z == 1) ? bk : bv;

    const float* base = x + ((size_t)b * CDIM) * HWSZ + hw;
    float s = 0.f, s2 = 0.f;
    for (int c = 0; c < CDIM; ++c) {
        float val = base[(size_t)c * HWSZ];
        s += val; s2 += val * val;
    }
    float mu = s * (1.f / CDIM);
    float var = s2 * (1.f / CDIM) - mu * mu;
    float rstd = rsqrtf(var + 1e-5f);
    half_t* o = out + (size_t)z * CDIM * PCOLS + (size_t)p * CDIM;
#pragma unroll 1
    for (int c0 = 0; c0 < CDIM; c0 += 8) {
        h8 vv;
#pragma unroll
        for (int i = 0; i < 8; ++i) {
            float val = base[(size_t)(c0 + i) * HWSZ];
            vv[i] = (half_t)((val - mu) * rstd * w[c0 + i] + bb[c0 + i]);
        }
        *(h8*)(o + c0) = vv;
    }
}

// ---------------------------------------------------------------------------
// 2) WMMA GEMM with TDM-staged activations:
//    Y[M,N] = W16[M,K] * Xt[N,K]^T + bias    (K=384)
//    A fragments direct from global (f16 weights, L2-resident).
//    B tiles (128 rows x 32 K) DMA'd by the Tensor Data Mover into LDS,
//    double-buffered on TENSORcnt -> tile arrives already in fragment order.
//    mode 0: Y f16 [M][N];  mode 1: OutAdd fp32 BCHW += result
// ---------------------------------------------------------------------------
__global__ __launch_bounds__(256) void gemm_wmma_tdm(
    const half_t* __restrict__ W16, const float* __restrict__ bias,
    const half_t* __restrict__ Xt, half_t* __restrict__ Y,
    float* __restrict__ OutAdd, int N, int mode)
{
    __shared__ __align__(16) half_t sBT[2][128 * 32];   // [n][k], 8KB per buffer

    int tid = threadIdx.x;
    int lane = tid & 31, wid = tid >> 5;
    int wr = wid >> 2, wc = wid & 3;
    int nBase = blockIdx.x * 128;
    int mBase = blockIdx.y * 128;

    unsigned int lds0 = (unsigned int)(uintptr_t)&sBT[0][0];
    const half_t* xtile = Xt + (size_t)nBase * CDIM;

    if (wid == 0)
        tdm_load_2d(lds0, xtile, 32, 128, CDIM, (unsigned)(N - nBase), CDIM);

    v8f acc[4][2] = {};

    for (int s = 0; s < 12; ++s) {
        int kk = s * 32;
        if (wid == 0) {
            if (s < 11) {
                tdm_load_2d(lds0 + ((s + 1) & 1) * (128 * 32 * 2), xtile + kk + 32,
                            32, 128, CDIM, (unsigned)(N - nBase), CDIM);
                __builtin_amdgcn_s_wait_tensorcnt(1);
            } else {
                __builtin_amdgcn_s_wait_tensorcnt(0);
            }
        }
        __syncthreads();

        const half_t* bufp = &sBT[s & 1][0];
        v16h af[4], bf[2];
#pragma unroll
        for (int i = 0; i < 4; ++i)
            af[i] = load_frag_a(W16 + (size_t)(mBase + wr * 64 + i * 16) * CDIM + kk, CDIM, lane);
#pragma unroll
        for (int j = 0; j < 2; ++j)
            bf[j] = load_frag_bt(bufp + (wc * 32 + j * 16) * 32, 32, lane);
#pragma unroll
        for (int i = 0; i < 4; ++i)
#pragma unroll
            for (int j = 0; j < 2; ++j) acc[i][j] = wmma_f16(af[i], bf[j], acc[i][j]);
        __syncthreads();
    }

    int cn = lane & 15, mo = (lane >> 4) * 8;
#pragma unroll
    for (int i = 0; i < 4; ++i) {
#pragma unroll
        for (int j = 0; j < 2; ++j) {
#pragma unroll
            for (int r = 0; r < 8; ++r) {
                int gm = mBase + wr * 64 + i * 16 + mo + r;
                int gn = nBase + wc * 32 + j * 16 + cn;
                float val = acc[i][j][r] + bias[gm];
                if (mode == 0) {
                    Y[(size_t)gm * N + gn] = (half_t)val;
                } else {
                    int b = gn / HWSZ, hw = gn % HWSZ;
                    size_t idx = ((size_t)b * CDIM + gm) * HWSZ + hw;
                    OutAdd[idx] += val;   // FFN residual accumulate
                }
            }
        }
    }
}

// ---------------------------------------------------------------------------
// 3) Depthwise 3x3 (f16 in [ch][P] -> f16 out [ch][P]), zero padded
// ---------------------------------------------------------------------------
__global__ __launch_bounds__(256) void dw3x3_f16(
    const half_t* __restrict__ in, const float* __restrict__ wdw,
    const float* __restrict__ bdw, half_t* __restrict__ out)
{
    long gid = (long)blockIdx.x * 256 + threadIdx.x;
    int ch = (int)(gid / PCOLS);
    int p = (int)(gid % PCOLS);
    int b = p / HWSZ, hw = p % HWSZ;
    int y = hw / WDIM, x = hw % WDIM;
    const half_t* base = in + (size_t)ch * PCOLS + (size_t)b * HWSZ;
    const float* w = wdw + ch * 9;
    float acc = bdw[ch];
#pragma unroll
    for (int ky = 0; ky < 3; ++ky) {
        int yy = y + ky - 1;
        if (yy < 0 || yy >= HDIM) continue;
#pragma unroll
        for (int kx = 0; kx < 3; ++kx) {
            int xx = x + kx - 1;
            if (xx < 0 || xx >= WDIM) continue;
            acc += w[ky * 3 + kx] * (float)base[yy * WDIM + xx];
        }
    }
    out[gid] = (half_t)acc;
}

// ---------------------------------------------------------------------------
// 4) Row L2 norms for q-heads (z=0) and k-heads (z=1)
// ---------------------------------------------------------------------------
__global__ __launch_bounds__(256) void row_norm(const half_t* __restrict__ buf,
                                                float* __restrict__ norms)
{
    __shared__ float red[256];
    int r = blockIdx.x;              // b*CDIM + c
    int z = blockIdx.y;
    int b = r / CDIM, c = r % CDIM;
    const half_t* base = buf + ((size_t)(z * CDIM + c)) * PCOLS + (size_t)b * HWSZ;
    float s = 0.f;
    for (int i = threadIdx.x; i < HWSZ; i += 256) {
        float v = (float)base[i];
        s += v * v;
    }
    red[threadIdx.x] = s;
    __syncthreads();
    for (int off = 128; off > 0; off >>= 1) {
        if (threadIdx.x < off) red[threadIdx.x] += red[threadIdx.x + off];
        __syncthreads();
    }
    if (threadIdx.x == 0)
        norms[z * (BATCH * CDIM) + r] = fmaxf(sqrtf(red[0]), 1e-12f);
}

// ---------------------------------------------------------------------------
// 5) Gram (q k^T) via WMMA over K=36864 + softmax -> attn16 [16][48][64] f16
// ---------------------------------------------------------------------------
__global__ __launch_bounds__(256) void attn_gram(
    const half_t* __restrict__ buf, const float* __restrict__ norms,
    const float* __restrict__ temperature, half_t* __restrict__ attn16)
{
    __shared__ float sS[HCH * HCH];
    int bh = blockIdx.x;
    int b = bh >> 3, h = bh & 7;
    int tid = threadIdx.x, lane = tid & 31, wid = tid >> 5;

    for (int i = tid; i < HCH * HCH; i += 256) sS[i] = 0.f;
    __syncthreads();

    const half_t* qBase = buf + ((size_t)(h * HCH)) * PCOLS + (size_t)b * HWSZ;
    const half_t* kBase = buf + ((size_t)(CDIM + h * HCH)) * PCOLS + (size_t)b * HWSZ;

    v8f acc[3][3] = {};
    const int span = HWSZ / 8;       // 4608
    int n0 = wid * span;
    for (int n = n0; n < n0 + span; n += 32) {
        v16h af[3], bf[3];
#pragma unroll
        for (int i = 0; i < 3; ++i) af[i] = load_frag_a(qBase + (size_t)(i * 16) * PCOLS + n, PCOLS, lane);
#pragma unroll
        for (int j = 0; j < 3; ++j) bf[j] = load_frag_bt(kBase + (size_t)(j * 16) * PCOLS + n, PCOLS, lane);
#pragma unroll
        for (int i = 0; i < 3; ++i)
#pragma unroll
            for (int j = 0; j < 3; ++j) acc[i][j] = wmma_f16(af[i], bf[j], acc[i][j]);
    }

    int cn = lane & 15, mo = (lane >> 4) * 8;
#pragma unroll
    for (int i = 0; i < 3; ++i)
#pragma unroll
        for (int j = 0; j < 3; ++j)
#pragma unroll
            for (int r = 0; r < 8; ++r)
                atomicAdd(&sS[(i * 16 + mo + r) * HCH + j * 16 + cn], acc[i][j][r]);
    __syncthreads();

    if (tid < HCH) {
        int c = tid;
        float qn = norms[b * CDIM + h * HCH + c];
        float t = temperature[h];
        float mx = -1e30f;
        for (int d = 0; d < HCH; ++d) {
            float kn = norms[BATCH * CDIM + b * CDIM + h * HCH + d];
            float v = sS[c * HCH + d] / (qn * kn) * t;
            mx = fmaxf(mx, v);
        }
        float sum = 0.f;
        for (int d = 0; d < HCH; ++d) {
            float kn = norms[BATCH * CDIM + b * CDIM + h * HCH + d];
            float v = sS[c * HCH + d] / (qn * kn) * t;
            sum += expf(v - mx);
        }
        float inv = 1.f / sum;
        half_t* o = attn16 + ((size_t)bh * HCH + c) * 64;
        for (int d = 0; d < HCH; ++d) {
            float kn = norms[BATCH * CDIM + b * CDIM + h * HCH + d];
            float v = sS[c * HCH + d] / (qn * kn) * t;
            o[d] = (half_t)(expf(v - mx) * inv);
        }
        for (int d = HCH; d < 64; ++d) o[d] = (half_t)0.f;   // zero pad K to 64
    }
}

// ---------------------------------------------------------------------------
// 6) out = attn @ V via WMMA.  M=48 (3 tiles), K=64 (2 steps), N tile = 256.
// ---------------------------------------------------------------------------
__global__ __launch_bounds__(256) void attn_apply(
    const half_t* __restrict__ attn16, const half_t* __restrict__ buf,
    half_t* __restrict__ outv)
{
    __shared__ __align__(16) half_t sVT[256 * 64];   // [n][k]
    int bh = blockIdx.y;
    int b = bh >> 3, h = bh & 7;
    int nBase = blockIdx.x * 256;
    int tid = threadIdx.x, lane = tid & 31, wid = tid >> 5;

    const half_t* vBase = buf + ((size_t)(2 * CDIM + h * HCH)) * PCOLS + (size_t)b * HWSZ + nBase;
#pragma unroll
    for (int c = 0; c < 8; ++c) {
        int id = c * 256 + tid;
        int k = id >> 5;
        int nn = (id & 31) * 8;
        h8 v = *(const h8*)(vBase + (size_t)k * PCOLS + nn);
#pragma unroll
        for (int i = 0; i < 8; ++i) sVT[(nn + i) * 64 + k] = v[i];
    }
    __syncthreads();

    const half_t* aptr = attn16 + (size_t)bh * HCH * 64;
    v8f acc[3][2] = {};
#pragma unroll
    for (int ks = 0; ks < 2; ++ks) {
        v16h af[3], bf[2];
#pragma unroll
        for (int i = 0; i < 3; ++i) af[i] = load_frag_a(aptr + (i * 16) * 64 + ks * 32, 64, lane);
#pragma unroll
        for (int j = 0; j < 2; ++j) bf[j] = load_frag_bt(&sVT[(wid * 32 + j * 16) * 64 + ks * 32], 64, lane);
#pragma unroll
        for (int i = 0; i < 3; ++i)
#pragma unroll
            for (int j = 0; j < 2; ++j) acc[i][j] = wmma_f16(af[i], bf[j], acc[i][j]);
    }

    int cn = lane & 15, mo = (lane >> 4) * 8;
#pragma unroll
    for (int i = 0; i < 3; ++i)
#pragma unroll
        for (int j = 0; j < 2; ++j)
#pragma unroll
            for (int r = 0; r < 8; ++r) {
                int ch = h * HCH + i * 16 + mo + r;
                int col = b * HWSZ + nBase + wid * 32 + j * 16 + cn;
                outv[(size_t)ch * PCOLS + col] = (half_t)acc[i][j][r];
            }
}

// ---------------------------------------------------------------------------
// 7) dwconv3x3(out) + b_proj + residual v  -> d_out (fp32 BCHW)
// ---------------------------------------------------------------------------
__global__ __launch_bounds__(256) void dw_proj_res(
    const half_t* __restrict__ in, const float* __restrict__ w,
    const float* __restrict__ bias, const float* __restrict__ vres,
    float* __restrict__ out)
{
    long gid = (long)blockIdx.x * 256 + threadIdx.x;
    int ch = (int)(gid / PCOLS);
    int p = (int)(gid % PCOLS);
    int b = p / HWSZ, hw = p % HWSZ;
    int y = hw / WDIM, x = hw % WDIM;
    const half_t* base = in + (size_t)ch * PCOLS + (size_t)b * HWSZ;
    const float* wc = w + ch * 9;
    float acc = bias[ch];
#pragma unroll
    for (int ky = 0; ky < 3; ++ky) {
        int yy = y + ky - 1;
        if (yy < 0 || yy >= HDIM) continue;
#pragma unroll
        for (int kx = 0; kx < 3; ++kx) {
            int xx = x + kx - 1;
            if (xx < 0 || xx >= WDIM) continue;
            acc += wc[ky * 3 + kx] * (float)base[yy * WDIM + xx];
        }
    }
    size_t oidx = ((size_t)b * CDIM + ch) * HWSZ + hw;
    out[oidx] = acc + vres[oidx];
}

// ---------------------------------------------------------------------------
// 8) FFN part 1: relu(dwconv3x3(v1)) -> f16 K-major [P][C]
//    thread handles 8 channels of one pixel -> 16B vector store
// ---------------------------------------------------------------------------
__global__ __launch_bounds__(256) void dw_ffn1_relu(
    const float* __restrict__ v1, const float* __restrict__ w,
    const float* __restrict__ bias, half_t* __restrict__ outT)
{
    long gid = (long)blockIdx.x * 256 + threadIdx.x;   // over P * 48 groups
    int p = (int)(gid / (CDIM / 8));
    int cg = (int)(gid % (CDIM / 8));
    int ch0 = cg * 8;
    int b = p / HWSZ, hw = p % HWSZ;
    int y = hw / WDIM, x = hw % WDIM;
    h8 ov;
#pragma unroll
    for (int j = 0; j < 8; ++j) {
        int ch = ch0 + j;
        const float* base = v1 + ((size_t)b * CDIM + ch) * HWSZ;
        const float* wc = w + ch * 9;
        float acc = bias[ch];
#pragma unroll
        for (int ky = 0; ky < 3; ++ky) {
            int yy = y + ky - 1;
            if (yy < 0 || yy >= HDIM) continue;
#pragma unroll
            for (int kx = 0; kx < 3; ++kx) {
                int xx = x + kx - 1;
                if (xx < 0 || xx >= WDIM) continue;
                acc += wc[ky * 3 + kx] * base[yy * WDIM + xx];
            }
        }
        ov[j] = (half_t)fmaxf(acc, 0.f);
    }
    *(h8*)(outT + (size_t)p * CDIM + ch0) = ov;
}

// ---------------------------------------------------------------------------
extern "C" void kernel_launch(void* const* d_in, const int* in_sizes, int n_in,
                              void* d_out, int out_size, void* d_ws, size_t ws_size,
                              hipStream_t stream) {
    const float* q      = (const float*)d_in[0];
    const float* k      = (const float*)d_in[1];
    const float* v      = (const float*)d_in[2];
    const float* ln_q_w = (const float*)d_in[3];
    const float* ln_q_b = (const float*)d_in[4];
    const float* ln_k_w = (const float*)d_in[5];
    const float* ln_k_b = (const float*)d_in[6];
    const float* ln_v_w = (const float*)d_in[7];
    const float* ln_v_b = (const float*)d_in[8];
    const float* wq     = (const float*)d_in[9];
    const float* bq     = (const float*)d_in[10];
    const float* wk     = (const float*)d_in[11];
    const float* bk     = (const float*)d_in[12];
    const float* wv     = (const float*)d_in[13];
    const float* bv     = (const float*)d_in[14];
    const float* w_dw   = (const float*)d_in[15];
    const float* b_dw   = (const float*)d_in[16];
    const float* temp   = (const float*)d_in[17];
    const float* w_proj = (const float*)d_in[18];
    const float* b_proj = (const float*)d_in[19];
    const float* w_ffn1 = (const float*)d_in[20];
    const float* b_ffn1 = (const float*)d_in[21];
    const float* w_ffn2 = (const float*)d_in[22];
    const float* b_ffn2 = (const float*)d_in[23];
    float* outF = (float*)d_out;

    const size_t P = PCOLS, C = CDIM;
    char* ws = (char*)d_ws;
    auto align256 = [](size_t x) { return (x + 255) & ~(size_t)255; };

    size_t off = 0;
    half_t* R1 = (half_t*)(ws + off);     // (3C+16) x P halves: LN K-major [3][P][C], later qkv [3C][P]
    off = align256(off + (3 * C + 16) * P * sizeof(half_t));
    half_t* R2 = (half_t*)(ws + off);     // 3C x P halves: conv1x1 out [3C][P], later FFN act [P][C]
    off = align256(off + 3 * C * P * sizeof(half_t));
    half_t* bufC = (half_t*)(ws + off);   // C x P  (attn@v out, [C][P])
    off = align256(off + C * P * sizeof(half_t));
    half_t* attn16 = (half_t*)(ws + off); // 16 x 48 x 64
    off = align256(off + 16 * HCH * 64 * sizeof(half_t));
    float* norms = (float*)(ws + off);    // 2 x (B*C)
    off = align256(off + 2 * BATCH * C * sizeof(float));
    half_t* W16 = (half_t*)(ws + off);    // 4 x 384 x 384 f16 weights
    off = align256(off + 4 * C * C * sizeof(half_t));

    // zero the 16 slack rows of R1 (read as padded K rows in attn_apply)
    hipMemsetAsync(R1 + 3 * C * P, 0, 16 * P * sizeof(half_t), stream);

    // 0) f16 weights
    conv_w16<<<dim3((unsigned)(4 * C * C / 256)), 256, 0, stream>>>(wq, wk, wv, w_ffn2, W16);

    // 1) LayerNorms -> f16 K-major
    ln_kernel<<<dim3((3 * P) / 256), 256, 0, stream>>>(
        q, k, v, ln_q_w, ln_q_b, ln_k_w, ln_k_b, ln_v_w, ln_v_b, R1);

    // 2) 1x1 convs as TDM-fed WMMA GEMMs  (R1 K-major -> R2 [ch][P])
    dim3 gg(P / 128, C / 128);
    gemm_wmma_tdm<<<gg, 256, 0, stream>>>(W16,             bq, R1,             R2,             nullptr, (int)P, 0);
    gemm_wmma_tdm<<<gg, 256, 0, stream>>>(W16 + C * C,     bk, R1 + C * P,     R2 + C * P,     nullptr, (int)P, 0);
    gemm_wmma_tdm<<<gg, 256, 0, stream>>>(W16 + 2 * C * C, bv, R1 + 2 * C * P, R2 + 2 * C * P, nullptr, (int)P, 0);

    // 3) joint depthwise 3x3 over 3C channels (R2 -> R1, now [3C][P])
    dw3x3_f16<<<dim3((unsigned)((3 * C * P) / 256)), 256, 0, stream>>>(R2, w_dw, b_dw, R1);

    // 4) L2 norms of q/k head rows
    row_norm<<<dim3(BATCH * C, 2), 256, 0, stream>>>(R1, norms);

    // 5) gram + softmax per (b,h)
    attn_gram<<<dim3(BATCH * NHEADS), 256, 0, stream>>>(R1, norms, temp, attn16);

    // 6) attn @ V
    attn_apply<<<dim3(HWSZ / 256, BATCH * NHEADS), 256, 0, stream>>>(attn16, R1, bufC);

    // 7) proj dwconv + residual -> d_out
    dw_proj_res<<<dim3((unsigned)((C * P) / 256)), 256, 0, stream>>>(bufC, w_proj, b_proj, v, outF);

    // 8) FFN dwconv + relu -> f16 K-major (into R2)
    dw_ffn1_relu<<<dim3((unsigned)((P * (C / 8)) / 256)), 256, 0, stream>>>(outF, w_ffn1, b_ffn1, R2);

    // 9) FFN 1x1 GEMM, += into d_out
    gemm_wmma_tdm<<<gg, 256, 0, stream>>>(W16 + 3 * C * C, b_ffn2, R2, nullptr, outF, (int)P, 1);
}
